// Attention_85770496901374
// MI455X (gfx1250) — matrix operand order repaired and checked by reading, not compiled, not run on hardware
//
#include <hip/hip_runtime.h>

// ---------------- problem constants ----------------
#define B_   2
#define S_   2048
#define E_   1024
#define H_   16
#define D_   64
#define ATTN_SCALE 0.03125f            // 1/sqrt(1024)
#define NEG_LOG2_ROPE_OVER_D (-0.20762050593046f)  // -log2(10000)/64

// ---------------- WMMA types ----------------
typedef __attribute__((ext_vector_type(16))) __bf16 v16bf;
typedef __attribute__((ext_vector_type(8)))  float  v8f;
typedef __attribute__((ext_vector_type(4)))  unsigned int v4u;

union FragB {            // 16 bf16 = 32 bytes, trivially constructible
    v16bf h;
    v4u   q[2];
    unsigned int u[8];
};

__device__ inline unsigned short f2bf(float f) {
    unsigned int u = __float_as_uint(f);
    u += 0x7FFFu + ((u >> 16) & 1u);     // round-to-nearest-even
    return (unsigned short)(u >> 16);
}
__device__ inline unsigned int pack2(unsigned short lo, unsigned short hi) {
    return (unsigned int)lo | ((unsigned int)hi << 16);
}

__device__ inline v8f wmma_bf16(const FragB& a, const FragB& b, v8f c) {
    return __builtin_amdgcn_wmma_f32_16x16x32_bf16(
        false, a.h, false, b.h, (short)0, c, false, false);
}

// Fragment loads from precomputed per-lane base pointers (offsets fold to
// instruction ioffset under unrolling).
// A (16x32): base = row_base + (lane<16 ? 0 : 8); groups at +0 and +16.
__device__ inline FragB load_a(const unsigned short* p) {
    FragB f;
    f.q[0] = *(const v4u*)(p);
    f.q[1] = *(const v4u*)(p + 16);
    return f;
}
// B (32x16 from transposed storage): base = col_base + (lane<16 ? 0 : 8);
// groups at +0 and +16.
__device__ inline FragB load_b(const unsigned short* p) {
    FragB f;
    f.q[0] = *(const v4u*)(p);
    f.q[1] = *(const v4u*)(p + 16);
    return f;
}

// Online-softmax update for one 16-query tile held transposed (lane = query).
__device__ inline float softmax_update(const v8f& st0, const v8f& st1,
                                       int key0, int qg, int roff,
                                       float& m_run, float& l_run, FragB& pa) {
    float s0v[8], s1v[8];
    float mx = -1e30f;
#pragma unroll
    for (int r = 0; r < 8; ++r) {
        int kg0 = key0 + roff + r;
        float v0 = st0[r] * ATTN_SCALE; if (kg0      > qg) v0 = -1e30f;
        float v1 = st1[r] * ATTN_SCALE; if (kg0 + 16 > qg) v1 = -1e30f;
        s0v[r] = v0; s1v[r] = v1;
        mx = fmaxf(mx, fmaxf(v0, v1));
    }
    mx = fmaxf(mx, __shfl_xor(mx, 16, 32));        // other 16 keys of row
    float mnew  = fmaxf(m_run, mx);
    float alpha = __expf(m_run - mnew);
    m_run = mnew;
    float sum = 0.0f;
#pragma unroll
    for (int j = 0; j < 4; ++j) {
        float p0a = __expf(s0v[2*j]     - mnew);
        float p0b = __expf(s0v[2*j + 1] - mnew);
        float p1a = __expf(s1v[2*j]     - mnew);
        float p1b = __expf(s1v[2*j + 1] - mnew);
        sum += p0a + p0b + p1a + p1b;
        pa.u[j]     = pack2(f2bf(p0a), f2bf(p0b));
        pa.u[4 + j] = pack2(f2bf(p1a), f2bf(p1b));
    }
    sum += __shfl_xor(sum, 16, 32);
    l_run = l_run * alpha + sum;
    return alpha;
}

// ---------------- kernel 1: f32 -> bf16 elementwise ----------------
__global__ void cvt_bf_kernel(const float* __restrict__ in,
                              unsigned short* __restrict__ out, int n) {
    int i = blockIdx.x * blockDim.x + threadIdx.x;
    if (i < n) out[i] = f2bf(in[i]);
}

// ---------------- kernel 2: weight transpose+convert (LDS tiled) ----------------
__global__ void cvt_wT_kernel(const float* __restrict__ w,
                              unsigned short* __restrict__ wT) {
    __shared__ float t[16][17];
    int bx = blockIdx.x * 16;   // k range
    int by = blockIdx.y * 16;   // n range
    t[threadIdx.y][threadIdx.x] = w[(size_t)(bx + threadIdx.y) * E_ + by + threadIdx.x];
    __syncthreads();
    wT[(size_t)(by + threadIdx.y) * E_ + bx + threadIdx.x] = f2bf(t[threadIdx.x][threadIdx.y]);
}

// ---------------- kernel 3: fused QKV projection + RoPE ----------------
// 32x64 per wave: 2 A-frags x 4 B-frags -> 8 WMMAs / k-step.
// grid (128, 4, 3) block (32,4); z: 0=Q 1=K 2=V.
__global__ void proj_qkv_kernel(const unsigned short* __restrict__ xbf,
                                const unsigned short* __restrict__ wT,
                                unsigned short* __restrict__ qbf,
                                unsigned short* __restrict__ kbf,
                                unsigned short* __restrict__ vT) {
    int lane   = threadIdx.x;
    int mt     = blockIdx.x;                             // 0..127, 32 rows each
    int nstrip = blockIdx.y * blockDim.y + threadIdx.y;  // 0..15, 64 cols each
    int mode   = blockIdx.z;
    const unsigned short* w = wT + (size_t)mode * E_ * E_;

    int row0 = mt * 32 + (lane & 15);
    int col0 = nstrip * 64 + (lane & 15);

    // loop-invariant per-lane base pointers
    const unsigned short* pa0 = xbf + (size_t)row0 * E_ + ((lane < 16) ? 0 : 8);
    const unsigned short* pa1 = pa0 + (size_t)16 * E_;
    const unsigned short* pb0 = w + (size_t)col0 * E_ + ((lane < 16) ? 0 : 8);

    v8f c[2][4] = {};
#pragma unroll 8
    for (int kk = 0; kk < E_; kk += 32) {
        // unconditional speculative prefetch (stays inside workspace; no guard
        // branch in the hot loop)
        __builtin_prefetch(pa0 + kk + 256, 0, 3);
        __builtin_prefetch(pb0 + kk + 256, 0, 3);
        FragB a0 = load_a(pa0 + kk);
        FragB a1 = load_a(pa1 + kk);
        FragB b[4];
#pragma unroll
        for (int nt = 0; nt < 4; ++nt)
            b[nt] = load_b(pb0 + (size_t)nt * 16 * E_ + kk);
#pragma unroll
        for (int nt = 0; nt < 4; ++nt) {
            c[0][nt] = wmma_bf16(a0, b[nt], c[0][nt]);
            c[1][nt] = wmma_bf16(a1, b[nt], c[1][nt]);
        }
    }

    int b_idx  = mt >> 6;             // (mt*32)/2048
    int s_base = (mt & 63) * 32;
    int roff   = (lane < 16) ? 0 : 8;
    int h      = nstrip;              // 64-wide strip == one head

    if (mode < 2) {
        unsigned short* dstb =
            ((mode == 0) ? qbf : kbf) + ((size_t)(b_idx * H_ + h) * S_) * D_;
#pragma unroll
        for (int nt = 0; nt < 4; ++nt) {
            int d = nt * 16 + (lane & 15);
            float inv_freq = exp2f((float)(d & ~1) * NEG_LOG2_ROPE_OVER_D);
            float sgn = (d & 1) ? 1.0f : -1.0f;
#pragma unroll
            for (int i = 0; i < 2; ++i) {
#pragma unroll
                for (int r = 0; r < 8; ++r) {
                    int s = s_base + i * 16 + roff + r;
                    float ang = (float)s * inv_freq;
                    float cs = __cosf(ang), sn = __sinf(ang);
                    float val  = c[i][nt][r];
                    float part = __shfl_xor(val, 1, 32);
                    dstb[(size_t)s * D_ + d] = f2bf(val * cs + sgn * part * sn);
                }
            }
        }
    } else {
#pragma unroll
        for (int nt = 0; nt < 4; ++nt) {
            int d = nt * 16 + (lane & 15);
#pragma unroll
            for (int i = 0; i < 2; ++i) {
                unsigned short* dst = vT +
                    ((size_t)(b_idx * H_ + h) * D_ + d) * S_ +
                    s_base + i * 16 + roff;
                v4u pk;
                pk.x = pack2(f2bf(c[i][nt][0]), f2bf(c[i][nt][1]));
                pk.y = pack2(f2bf(c[i][nt][2]), f2bf(c[i][nt][3]));
                pk.z = pack2(f2bf(c[i][nt][4]), f2bf(c[i][nt][5]));
                pk.w = pack2(f2bf(c[i][nt][6]), f2bf(c[i][nt][7]));
                *(v4u*)dst = pk;
            }
        }
    }
}

// ---------------- kernel 4: flash attention (32 queries / wave) ----------------
// grid (S/32, B*H), block 32. S^T = K @ Q^T keeps exp'd scores in the bf16
// A-fragment layout for P @ V; K/V fragments shared by both 16-query tiles.
__global__ void flash_attn_kernel(const unsigned short* __restrict__ qbf,
                                  const unsigned short* __restrict__ kbf,
                                  const unsigned short* __restrict__ vT,
                                  unsigned short* __restrict__ attn) {
    int lane = threadIdx.x;
    int qt   = blockIdx.x;          // 0..63
    int bh   = blockIdx.y;          // 0..31
    int q0   = qt * 32;

    const unsigned short* qb = qbf + (size_t)bh * S_ * D_;
    const unsigned short* kb = kbf + (size_t)bh * S_ * D_;
    const unsigned short* vb = vT  + (size_t)bh * D_ * S_;

    FragB bqA[2], bqB[2];
    {
        const unsigned short* pq =
            qb + (size_t)(q0 + (lane & 15)) * D_ + ((lane < 16) ? 0 : 8);
        bqA[0] = load_b(pq);
        bqA[1] = load_b(pq + 32);
        bqB[0] = load_b(pq + 16 * D_);
        bqB[1] = load_b(pq + 16 * D_ + 32);
    }

    // per-lane base pointers advanced per key block
    const unsigned short* pk0 = kb + (size_t)(lane & 15) * D_ + ((lane < 16) ? 0 : 8);
    const unsigned short* pk1 = pk0 + (size_t)16 * D_;
    const unsigned short* pv0 = vb + (size_t)(lane & 15) * S_ + ((lane < 16) ? 0 : 8);

    v8f oA[4] = {}, oB[4] = {};
    float mA = -1e30f, lA = 0.0f, mB = -1e30f, lB = 0.0f;
    int qgA  = q0 + (lane & 15);
    int qgB  = qgA + 16;
    int roff = (lane < 16) ? 0 : 8;
    int kb_max = (q0 + 31) >> 5;

    for (int kblk = 0; kblk <= kb_max; ++kblk) {
        int key0 = kblk * 32;

        // ---- S^T tiles for both query tiles (K-frags shared) ----
        v8f stA0 = {}, stA1 = {}, stB0 = {}, stB1 = {};
#pragma unroll
        for (int ks = 0; ks < 2; ++ks) {
            FragB a0 = load_a(pk0 + ks * 32);
            FragB a1 = load_a(pk1 + ks * 32);
            stA0 = wmma_bf16(a0, bqA[ks], stA0);
            stA1 = wmma_bf16(a1, bqA[ks], stA1);
            stB0 = wmma_bf16(a0, bqB[ks], stB0);
            stB1 = wmma_bf16(a1, bqB[ks], stB1);
        }

        // ---- online softmax per query tile ----
        FragB paA, paB;
        float alphaA = softmax_update(stA0, stA1, key0, qgA, roff, mA, lA, paA);
        float alphaB = softmax_update(stB0, stB1, key0, qgB, roff, mB, lB, paB);

        // ---- rescale O accumulators (row r needs alpha of query m) ----
#pragma unroll
        for (int r = 0; r < 8; ++r) {
            float aAm = __shfl(alphaA, roff + r, 32);
            float aBm = __shfl(alphaB, roff + r, 32);
#pragma unroll
            for (int nt = 0; nt < 4; ++nt) {
                oA[nt][r] *= aAm;
                oB[nt][r] *= aBm;
            }
        }

        // ---- O += P @ V  (V-frags shared by both query tiles) ----
#pragma unroll
        for (int nt = 0; nt < 4; ++nt) {
            FragB bv = load_b(pv0 + (size_t)nt * 16 * S_);
            oA[nt] = wmma_bf16(paA, bv, oA[nt]);
            oB[nt] = wmma_bf16(paB, bv, oB[nt]);
        }

        pk0 += (size_t)32 * D_;
        pk1 += (size_t)32 * D_;
        pv0 += 32;
    }

    // ---- normalize and store to attn[b][s][e] (bf16) ----
    float liA = 1.0f / lA, liB = 1.0f / lB;
    int h = bh & 15, b_idx = bh >> 4;
#pragma unroll
    for (int r = 0; r < 8; ++r) {
        float lmA = __shfl(liA, roff + r, 32);
        float lmB = __shfl(liB, roff + r, 32);
        int sA = q0 + roff + r;
        int sB = sA + 16;
        unsigned short* dA = attn + ((size_t)(b_idx * S_ + sA)) * E_ + h * D_;
        unsigned short* dB = attn + ((size_t)(b_idx * S_ + sB)) * E_ + h * D_;
#pragma unroll
        for (int nt = 0; nt < 4; ++nt) {
            dA[nt * 16 + (lane & 15)] = f2bf(oA[nt][r] * lmA);
            dB[nt * 16 + (lane & 15)] = f2bf(oB[nt][r] * lmB);
        }
    }
}

// ---------------- kernel 5: output projection -> f32 ----------------
__global__ void out_proj_kernel(const unsigned short* __restrict__ attnbf,
                                const unsigned short* __restrict__ woT,
                                float* __restrict__ out) {
    int lane   = threadIdx.x;
    int mt     = blockIdx.x;                             // 0..127
    int nstrip = blockIdx.y * blockDim.y + threadIdx.y;  // 0..15
    int row0 = mt * 32 + (lane & 15);
    int col0 = nstrip * 64 + (lane & 15);

    const unsigned short* pa0 = attnbf + (size_t)row0 * E_ + ((lane < 16) ? 0 : 8);
    const unsigned short* pa1 = pa0 + (size_t)16 * E_;
    const unsigned short* pb0 = woT + (size_t)col0 * E_ + ((lane < 16) ? 0 : 8);

    v8f c[2][4] = {};
#pragma unroll 8
    for (int kk = 0; kk < E_; kk += 32) {
        __builtin_prefetch(pa0 + kk + 256, 0, 3);
        __builtin_prefetch(pb0 + kk + 256, 0, 3);
        FragB a0 = load_a(pa0 + kk);
        FragB a1 = load_a(pa1 + kk);
        FragB b[4];
#pragma unroll
        for (int nt = 0; nt < 4; ++nt)
            b[nt] = load_b(pb0 + (size_t)nt * 16 * E_ + kk);
#pragma unroll
        for (int nt = 0; nt < 4; ++nt) {
            c[0][nt] = wmma_bf16(a0, b[nt], c[0][nt]);
            c[1][nt] = wmma_bf16(a1, b[nt], c[1][nt]);
        }
    }
    int roff = (lane < 16) ? 0 : 8;
#pragma unroll
    for (int nt = 0; nt < 4; ++nt) {
        int col = nstrip * 64 + nt * 16 + (lane & 15);
#pragma unroll
        for (int i = 0; i < 2; ++i)
#pragma unroll
            for (int r = 0; r < 8; ++r)
                out[(size_t)(mt * 32 + i * 16 + roff + r) * E_ + col] = c[i][nt][r];
    }
}

// ---------------- host launcher ----------------
extern "C" void kernel_launch(void* const* d_in, const int* in_sizes, int n_in,
                              void* d_out, int out_size, void* d_ws, size_t ws_size,
                              hipStream_t stream) {
    const float* x  = (const float*)d_in[0];
    const float* Wq = (const float*)d_in[1];
    const float* Wk = (const float*)d_in[2];
    const float* Wv = (const float*)d_in[3];
    const float* Wo = (const float*)d_in[4];

    unsigned short* ws = (unsigned short*)d_ws;
    const size_t M1 = (size_t)1024 * 1024;
    unsigned short* xbf  = ws;                 // 4M bf16
    unsigned short* wT   = ws + 4 * M1;        // 4x 1M bf16 (q,k,v,o transposed)
    unsigned short* qbf  = ws + 8 * M1;        // 4M
    unsigned short* kbf  = ws + 12 * M1;       // 4M
    unsigned short* vT   = ws + 16 * M1;       // 4M (transposed [bh][d][s])
    unsigned short* attn = ws + 20 * M1;       // 4M

    const int n_x = B_ * S_ * E_;              // 4194304
    cvt_bf_kernel<<<(n_x + 255) / 256, 256, 0, stream>>>(x, xbf, n_x);

    dim3 tg(16, 16), tgrid(E_ / 16, E_ / 16);
    cvt_wT_kernel<<<tgrid, tg, 0, stream>>>(Wq, wT + 0 * M1);
    cvt_wT_kernel<<<tgrid, tg, 0, stream>>>(Wk, wT + 1 * M1);
    cvt_wT_kernel<<<tgrid, tg, 0, stream>>>(Wv, wT + 2 * M1);
    cvt_wT_kernel<<<tgrid, tg, 0, stream>>>(Wo, wT + 3 * M1);

    proj_qkv_kernel<<<dim3(128, 4, 3), dim3(32, 4), 0, stream>>>(
        xbf, wT, qbf, kbf, vT);

    flash_attn_kernel<<<dim3(S_ / 32, B_ * H_), dim3(32, 1), 0, stream>>>(
        qbf, kbf, vT, attn);

    out_proj_kernel<<<dim3(128, 4), dim3(32, 4), 0, stream>>>(
        attn, wT + 3 * M1, (float*)d_out);
}
